// RWKVTimeMixing_8675833938458
// MI455X (gfx1250) — matrix operand order repaired and checked by reading, not compiled, not run on hardware
//
#include <hip/hip_runtime.h>
#include <hip/hip_bf16.h>

// ---------------------------------------------------------------------------
// RWKV time-mixing for MI455X (gfx1250, wave32, WMMA).
// Heavy GEMMs use v_wmma_f32_16x16x32_bf16 (bf16 A/B, f32 accumulate).
// Elementwise math (sigmoid/exp/LN/recurrence) stays f32.
// ---------------------------------------------------------------------------

typedef __bf16 bf16_t;
typedef __attribute__((ext_vector_type(16))) __bf16 v16bf;
typedef __attribute__((ext_vector_type(8)))  __bf16 v8bf;
typedef __attribute__((ext_vector_type(8)))  float  v8f;

#define D_ATTN    2048
#define D_STATE   64
#define B_SZ      8
#define T_SZ      2048
#define N_ROWS    (B_SZ * T_SZ)      // 16384 rows
#define SHIFT_LEN 1024               // T * SHIFT_FRAC
#define LN_EPS    1e-5f

union FragBF { v16bf v; v8bf h[2]; };

__device__ __forceinline__ bf16_t f2bf(float f) {
    // round-to-nearest-even f32 -> bf16
    unsigned int u = __float_as_uint(f);
    unsigned int r = (u + 0x7FFFu + ((u >> 16) & 1u)) >> 16;
    union { unsigned short s; bf16_t b; } cv;
    cv.s = (unsigned short)r;
    return cv.b;
}

__device__ __forceinline__ float sigmoidf_(float x) {
    return 1.0f / (1.0f + expf(-x));
}

// A fragment (16x32 bf16, row-major source). contiguous k order of this code (B matches):
// lane<16 : M=lane,    VGPR0-7 = K 0..15
// lane>=16: M=lane-16, VGPR0-7 = K 16..31
__device__ __forceinline__ v16bf load_a_frag(const bf16_t* __restrict__ base,
                                             int lda, int row, int k, int lane) {
    const int half = lane >> 4;
    const bf16_t* p = base + (size_t)row * lda + k + half * 16;
    FragBF f;
    f.h[0] = *(const v8bf*)(p);        // K = k + half*16     .. +7
    f.h[1] = *(const v8bf*)(p + 8);    // K = k + half*16 + 8 .. +7
    return f.v;
}

// B fragment (32x16 bf16). Column n = W row n (natural weight layout):
// lane<16 : N=lane,    K 0..15 packed 2/VGPR ; lane>=16: N=lane-16, K 16..31
__device__ __forceinline__ v16bf load_b_frag(const bf16_t* __restrict__ wbase,
                                             int ldb, int col, int k, int lane) {
    const int half = lane >> 4;
    const bf16_t* p = wbase + (size_t)col * ldb + k + half * 16;
    return *(const v16bf*)p;           // 32B contiguous, 32B aligned
}

// ---------------------------------------------------------------------------
// Prep: roll+cast x to bf16, cast weights to bf16 (Wk/Wv concatenated 128 rows)
// ---------------------------------------------------------------------------
__global__ void prep_cast_kernel(const float* __restrict__ x,
                                 const float* __restrict__ wsh,
                                 const float* __restrict__ wk,
                                 const float* __restrict__ wv,
                                 const float* __restrict__ wo,
                                 bf16_t* __restrict__ xb,
                                 bf16_t* __restrict__ wshb,
                                 bf16_t* __restrict__ wkvb,
                                 bf16_t* __restrict__ wob) {
    size_t i0 = (size_t)blockIdx.x * blockDim.x + threadIdx.x;
    size_t stride = (size_t)gridDim.x * blockDim.x;
    const size_t NX = (size_t)N_ROWS * D_ATTN;
    for (size_t i = i0; i < NX; i += stride) {
        int d = (int)(i % D_ATTN);
        size_t n = i / D_ATTN;
        int t = (int)(n % T_SZ);
        int b = (int)(n / T_SZ);
        int ts = (t + SHIFT_LEN) & (T_SZ - 1);     // roll(x, -shift, axis=1)
        xb[i] = f2bf(x[((size_t)b * T_SZ + ts) * D_ATTN + d]);
    }
    const size_t NW = (size_t)D_ATTN * D_ATTN;
    for (size_t i = i0; i < NW; i += stride) wshb[i] = f2bf(wsh[i]);
    const size_t NK = (size_t)D_STATE * D_ATTN;
    for (size_t i = i0; i < NK; i += stride) {
        wkvb[i]      = f2bf(wk[i]);
        wkvb[NK + i] = f2bf(wv[i]);
    }
    const size_t NO = (size_t)D_ATTN * D_STATE;
    for (size_t i = i0; i < NO; i += stride) wob[i] = f2bf(wo[i]);
}

// ---------------------------------------------------------------------------
// GEMM1: x_shift = (x_cat @ W_shift^T) * g + x * (1-g)
// block = 8 waves (4m x 2n), wave tile 64x64, block tile 256x128
// ---------------------------------------------------------------------------
__global__ void __launch_bounds__(256)
gemm_shift_kernel(const bf16_t* __restrict__ xb, const bf16_t* __restrict__ wshb,
                  const float* __restrict__ x, const float* __restrict__ gate,
                  float* __restrict__ xshift) {
    const int lane = threadIdx.x & 31;
    const int wave = threadIdx.x >> 5;
    const int mbase = blockIdx.x * 256 + (wave & 3) * 64;
    const int nbase = blockIdx.y * 128 + (wave >> 2) * 64;
    v8f acc[4][4] = {};
    for (int kk = 0; kk < D_ATTN; kk += 32) {
        v16bf a[4], b[4];
#pragma unroll
        for (int mi = 0; mi < 4; mi++)
            a[mi] = load_a_frag(xb, D_ATTN, mbase + mi * 16 + (lane & 15), kk, lane);
#pragma unroll
        for (int ni = 0; ni < 4; ni++)
            b[ni] = load_b_frag(wshb, D_ATTN, nbase + ni * 16 + (lane & 15), kk, lane);
        if (kk + 32 < D_ATTN)   // hint next A chunk -> global_prefetch_b8
            __builtin_prefetch(xb + (size_t)(mbase + (lane & 15)) * D_ATTN + kk + 32, 0, 0);
#pragma unroll
        for (int mi = 0; mi < 4; mi++)
#pragma unroll
            for (int ni = 0; ni < 4; ni++)
                acc[mi][ni] = __builtin_amdgcn_wmma_f32_16x16x32_bf16(
                    false, a[mi], false, b[ni], (short)0, acc[mi][ni], false, false);
    }
    const int half = lane >> 4;
#pragma unroll
    for (int ni = 0; ni < 4; ni++) {
        const int col = nbase + ni * 16 + (lane & 15);
        const float gv = sigmoidf_(gate[col]);
#pragma unroll
        for (int mi = 0; mi < 4; mi++) {
#pragma unroll
            for (int r = 0; r < 8; r++) {
                const int row = mbase + mi * 16 + half * 8 + r;
                const size_t idx = (size_t)row * D_ATTN + col;
                xshift[idx] = acc[mi][ni][r] * gv + x[idx] * (1.0f - gv);
            }
        }
    }
}

// ---------------------------------------------------------------------------
// LayerNorm per row + cast to bf16
// ---------------------------------------------------------------------------
__global__ void __launch_bounds__(256)
ln_kernel(const float* __restrict__ xshift, const float* __restrict__ gamma,
          const float* __restrict__ beta, bf16_t* __restrict__ xnb) {
    __shared__ float s1[256], s2[256];
    const int row = blockIdx.x;
    const float* xr = xshift + (size_t)row * D_ATTN;
    float sum = 0.0f, sq = 0.0f;
    for (int d = threadIdx.x; d < D_ATTN; d += 256) {
        float v = xr[d]; sum += v; sq += v * v;
    }
    s1[threadIdx.x] = sum; s2[threadIdx.x] = sq;
    __syncthreads();
    for (int off = 128; off > 0; off >>= 1) {
        if (threadIdx.x < off) {
            s1[threadIdx.x] += s1[threadIdx.x + off];
            s2[threadIdx.x] += s2[threadIdx.x + off];
        }
        __syncthreads();
    }
    const float mu   = s1[0] * (1.0f / D_ATTN);
    const float var  = s2[0] * (1.0f / D_ATTN) - mu * mu;
    const float rstd = rsqrtf(var + LN_EPS);
    bf16_t* onb = xnb + (size_t)row * D_ATTN;
    for (int d = threadIdx.x; d < D_ATTN; d += 256)
        onb[d] = f2bf((xr[d] - mu) * rstd * gamma[d] + beta[d]);
}

// ---------------------------------------------------------------------------
// GEMM2: kv[n, 0:64]=k, kv[n,64:128]=v  (Wkv = concat(W_key, W_value) rows)
// block tile 256x128, grid (N/256, 1)
// ---------------------------------------------------------------------------
__global__ void __launch_bounds__(256)
gemm_kv_kernel(const bf16_t* __restrict__ xnb, const bf16_t* __restrict__ wkvb,
               float* __restrict__ kv) {
    const int lane = threadIdx.x & 31;
    const int wave = threadIdx.x >> 5;
    const int mbase = blockIdx.x * 256 + (wave & 3) * 64;
    const int nbase = (wave >> 2) * 64;
    v8f acc[4][4] = {};
    for (int kk = 0; kk < D_ATTN; kk += 32) {
        v16bf a[4], b[4];
#pragma unroll
        for (int mi = 0; mi < 4; mi++)
            a[mi] = load_a_frag(xnb, D_ATTN, mbase + mi * 16 + (lane & 15), kk, lane);
#pragma unroll
        for (int ni = 0; ni < 4; ni++)
            b[ni] = load_b_frag(wkvb, D_ATTN, nbase + ni * 16 + (lane & 15), kk, lane);
#pragma unroll
        for (int mi = 0; mi < 4; mi++)
#pragma unroll
            for (int ni = 0; ni < 4; ni++)
                acc[mi][ni] = __builtin_amdgcn_wmma_f32_16x16x32_bf16(
                    false, a[mi], false, b[ni], (short)0, acc[mi][ni], false, false);
    }
    const int half = lane >> 4;
#pragma unroll
    for (int ni = 0; ni < 4; ni++) {
        const int col = nbase + ni * 16 + (lane & 15);
#pragma unroll
        for (int mi = 0; mi < 4; mi++)
#pragma unroll
            for (int r = 0; r < 8; r++) {
                const int row = mbase + mi * 16 + half * 8 + r;
                kv[(size_t)row * 128 + col] = acc[mi][ni][r];
            }
    }
}

// ---------------------------------------------------------------------------
// WKV recurrence: 512 independent (b,s) sequences; writes bf16 states and
// last_state into the tail of d_out.
// ---------------------------------------------------------------------------
__global__ void recurrence_kernel(const float* __restrict__ kv,
                                  const float* __restrict__ time_decay,
                                  const float* __restrict__ time_first,
                                  bf16_t* __restrict__ statesb,
                                  float* __restrict__ last_state) {
    const int tid = blockIdx.x * blockDim.x + threadIdx.x;
    if (tid >= B_SZ * D_STATE) return;
    const int b = tid / D_STATE, s = tid % D_STATE;
    const float w = expf(time_decay[s]);
    const float u = expf(time_first[s]);
    const float* kvb = kv + (size_t)b * T_SZ * 128;
    bf16_t* sb = statesb + (size_t)b * T_SZ * D_STATE;
    float st = 0.0f;
    for (int t = 0; t < T_SZ; t++) {
        const float kval = kvb[(size_t)t * 128 + s];
        const float vval = kvb[(size_t)t * 128 + 64 + s];
        const float wkv = expf(-u * (1.0f / (1.0f + expf(-kval)))) * vval;
        st = st * w + wkv;
        sb[(size_t)t * D_STATE + s] = f2bf(st);
    }
    last_state[tid] = st;
}

// ---------------------------------------------------------------------------
// GEMM3: output = states @ W_output^T  (K=64, two WMMA steps)
// ---------------------------------------------------------------------------
__global__ void __launch_bounds__(256)
gemm_out_kernel(const bf16_t* __restrict__ statesb, const bf16_t* __restrict__ wob,
                float* __restrict__ out) {
    const int lane = threadIdx.x & 31;
    const int wave = threadIdx.x >> 5;
    const int mbase = blockIdx.x * 256 + (wave & 3) * 64;
    const int nbase = blockIdx.y * 128 + (wave >> 2) * 64;
    v8f acc[4][4] = {};
#pragma unroll
    for (int kk = 0; kk < D_STATE; kk += 32) {
        v16bf a[4], b[4];
#pragma unroll
        for (int mi = 0; mi < 4; mi++)
            a[mi] = load_a_frag(statesb, D_STATE, mbase + mi * 16 + (lane & 15), kk, lane);
#pragma unroll
        for (int ni = 0; ni < 4; ni++)
            b[ni] = load_b_frag(wob, D_STATE, nbase + ni * 16 + (lane & 15), kk, lane);
#pragma unroll
        for (int mi = 0; mi < 4; mi++)
#pragma unroll
            for (int ni = 0; ni < 4; ni++)
                acc[mi][ni] = __builtin_amdgcn_wmma_f32_16x16x32_bf16(
                    false, a[mi], false, b[ni], (short)0, acc[mi][ni], false, false);
    }
    const int half = lane >> 4;
#pragma unroll
    for (int ni = 0; ni < 4; ni++) {
        const int col = nbase + ni * 16 + (lane & 15);
#pragma unroll
        for (int mi = 0; mi < 4; mi++)
#pragma unroll
            for (int r = 0; r < 8; r++) {
                const int row = mbase + mi * 16 + half * 8 + r;
                out[(size_t)row * D_ATTN + col] = acc[mi][ni][r];
            }
    }
}

// ---------------------------------------------------------------------------
// Workspace layout (bytes)
// ---------------------------------------------------------------------------
static const size_t OFF_XB     = 0;                                          // bf16 N x D (rolled x; reused for x_norm)
static const size_t OFF_WSH    = OFF_XB     + (size_t)N_ROWS * D_ATTN * 2;   // bf16 D x D
static const size_t OFF_WKV    = OFF_WSH    + (size_t)D_ATTN * D_ATTN * 2;   // bf16 128 x D
static const size_t OFF_WOUT   = OFF_WKV    + (size_t)2 * D_STATE * D_ATTN * 2; // bf16 D x 64
static const size_t OFF_KV     = OFF_WOUT   + (size_t)D_ATTN * D_STATE * 2;  // f32  N x 128
static const size_t OFF_SB     = OFF_KV     + (size_t)N_ROWS * 128 * 4;      // bf16 N x 64
static const size_t OFF_XSHIFT = OFF_SB     + (size_t)N_ROWS * D_STATE * 2;  // f32  N x D

extern "C" void kernel_launch(void* const* d_in, const int* in_sizes, int n_in,
                              void* d_out, int out_size, void* d_ws, size_t ws_size,
                              hipStream_t stream) {
    const float* x          = (const float*)d_in[0];
    const float* W_key      = (const float*)d_in[1];
    const float* W_value    = (const float*)d_in[2];
    const float* W_output   = (const float*)d_in[3];
    const float* W_shift    = (const float*)d_in[4];
    const float* gate       = (const float*)d_in[5];
    const float* time_decay = (const float*)d_in[6];
    const float* time_first = (const float*)d_in[7];
    const float* ln_gamma   = (const float*)d_in[8];
    const float* ln_beta    = (const float*)d_in[9];
    float* out = (float*)d_out;

    char* ws = (char*)d_ws;
    bf16_t* xb      = (bf16_t*)(ws + OFF_XB);      // rolled x (bf16); later x_norm (bf16)
    bf16_t* wshb    = (bf16_t*)(ws + OFF_WSH);
    bf16_t* wkvb    = (bf16_t*)(ws + OFF_WKV);
    bf16_t* wob     = (bf16_t*)(ws + OFF_WOUT);
    float*  kv      = (float*) (ws + OFF_KV);
    bf16_t* statesb = (bf16_t*)(ws + OFF_SB);
    float*  xshift  = (float*) (ws + OFF_XSHIFT);

    // 1) casts + roll
    prep_cast_kernel<<<4096, 256, 0, stream>>>(x, W_shift, W_key, W_value, W_output,
                                               xb, wshb, wkvb, wob);
    // 2) token-shift GEMM + gate blend
    gemm_shift_kernel<<<dim3(N_ROWS / 256, D_ATTN / 128), 256, 0, stream>>>(
        xb, wshb, x, gate, xshift);
    // 3) LayerNorm (+ cast); overwrites xb region (rolled x is dead now)
    ln_kernel<<<N_ROWS, 256, 0, stream>>>(xshift, ln_gamma, ln_beta, xb);
    // 4) k/v GEMM (128 output cols)
    gemm_kv_kernel<<<dim3(N_ROWS / 256, 1), 256, 0, stream>>>(xb, wkvb, kv);
    // 5) recurrence; last_state -> tail of d_out
    recurrence_kernel<<<2, 256, 0, stream>>>(kv, time_decay, time_first, statesb,
                                             out + (size_t)N_ROWS * D_ATTN);
    // 6) output projection GEMM
    gemm_out_kernel<<<dim3(N_ROWS / 256, D_ATTN / 128), 256, 0, stream>>>(
        statesb, wob, out);
}
